// ConvLSTM2DCell_42236708388992
// MI455X (gfx1250) — matrix operand order repaired and checked by reading, not compiled, not run on hardware
//
#include <hip/hip_runtime.h>
#include <hip/hip_bf16.h>

typedef __attribute__((ext_vector_type(16))) __bf16 v16bf;
typedef __attribute__((ext_vector_type(8)))  float  v8f;

#define B_    16
#define CIN   64
#define CHID  128
#define CCOMB 192            // CIN + CHID
#define KTOT  1728           // CCOMB * 9, ordered k = (r*3+s)*192 + c
#define OC    512
#define NPIX  65536          // B*H*W
#define HP    66             // padded height (1-row halo)
#define WP    72             // padded width  (1-col halo + align pad)
#define CHST  (HP * WP)      // 4752 elements per channel plane

// ---------------------------------------------------------------------------
// gfx1250 async global->LDS copy (ASYNCcnt) + native tanh transcendental
// ---------------------------------------------------------------------------
__device__ __forceinline__ void async_lds_b128(void* lds, const void* gptr) {
    unsigned ldsoff = (unsigned)(unsigned long long)lds;
    asm volatile("global_load_async_to_lds_b128 %0, %1, off"
                 :: "v"(ldsoff), "v"(gptr) : "memory");
}
__device__ __forceinline__ void wait_asynccnt0() {
    asm volatile("s_wait_asynccnt 0x0" ::: "memory");
}
__device__ __forceinline__ float tanh_hw(float x) {
    float r;
    asm("v_tanh_f32 %0, %1" : "=v"(r) : "v"(x));
    return r;
}
__device__ __forceinline__ float sigmoid_hw(float x) {
    return fmaf(0.5f, tanh_hw(0.5f * x), 0.5f);
}

// ---------------------------------------------------------------------------
// Prep 1: weights f32 OIHW -> bf16 [oc][rs*192 + c]; grid (512,9), block 192
// ---------------------------------------------------------------------------
__global__ __launch_bounds__(192) void prep_weights(const float* __restrict__ w,
                                                    __bf16* __restrict__ wt) {
    int oc = blockIdx.x, rs = blockIdx.y, c = threadIdx.x;
    wt[oc * KTOT + rs * CCOMB + c] = (__bf16)w[oc * KTOT + c * 9 + rs];
}

// ---------------------------------------------------------------------------
// Prep 2: zero-padded comb[b][c][66][72] bf16 (halo = 0).
// grid = (19, 192, 16), block = 256
// ---------------------------------------------------------------------------
__global__ __launch_bounds__(256) void prep_comb(const float* __restrict__ x,
                                                 const float* __restrict__ h,
                                                 __bf16* __restrict__ combP) {
    int e = blockIdx.x * 256 + threadIdx.x;
    if (e >= CHST) return;
    int c = blockIdx.y, b = blockIdx.z;
    int yy = e / WP;
    int xx = e - yy * WP;
    int yi = yy - 1, xi = xx - 1;
    float v = 0.0f;
    if ((unsigned)yi < 64u && (unsigned)xi < 64u) {
        int pix = (yi << 6) + xi;
        v = (c < CIN) ? x[(b * CIN + c) * 4096 + pix]
                      : h[(b * CHID + (c - CIN)) * 4096 + pix];
    }
    combP[(b * CCOMB + c) * CHST + e] = (__bf16)v;
}

// ---------------------------------------------------------------------------
// Main: implicit-GEMM conv (bf16 WMMA) + fused LSTM gating.
// Pipeline unrolled x2: compile-time buffer indices, unconditional staging.
// ---------------------------------------------------------------------------
__global__ __launch_bounds__(256) void convlstm_wmma(
    const __bf16* __restrict__ wt,     // [OC][KTOT]
    const __bf16* __restrict__ combP,  // [B][192][66][72]
    const float*  __restrict__ c_cur,
    const float*  __restrict__ bias,
    float* __restrict__ h_next,
    float* __restrict__ c_next)
{
    __shared__ union {
        struct {
            __align__(16) __bf16 W[2][128][40];
            __align__(16) __bf16 P[2][64][40];
        } st;                                    // 30720 B
        float Sout[128][66];                     // 33792 B (post-loop overlay)
    } smem;

    const int tid  = threadIdx.x;
    const int lane = tid & 31;
    const int wv   = tid >> 5;
    const int g    = wv & 3;
    const int t    = wv >> 2;
    const int m    = lane & 15;
    const int hi   = lane >> 4;

    const int pixBase = blockIdx.x * 64;     // one row of one batch image
    const int chb     = blockIdx.y * 32;
    const int b       = pixBase >> 12;
    const int y       = (pixBase & 4095) >> 6;

    // ---- weight async-copy slots (2 per thread) ----
    int ocl0 = tid >> 2, ocl1 = ocl0 + 64;
    int k8   = (tid & 3) << 3;
    const __bf16* wg0 = wt + ((ocl0 >> 5) * CHID + chb + (ocl0 & 31)) * KTOT + k8;
    const __bf16* wg1 = wt + ((ocl1 >> 5) * CHID + chb + (ocl1 & 31)) * KTOT + k8;
    void* wl0[2] = { &smem.st.W[0][ocl0][k8], &smem.st.W[1][ocl0][k8] };
    void* wl1[2] = { &smem.st.W[0][ocl1][k8], &smem.st.W[1][ocl1][k8] };

    // ---- patch staging invariants ----
    const int pl   = tid & 63;
    const int kkb  = (tid >> 6) << 3;
    const int tOff = kkb * CHST + pl;

    auto patchBase = [&](int kc) -> const __bf16* {
        int chunk = kc >> 5;                 // 0..53
        int rs    = chunk / 6;               // 0..8 (SALU)
        int cBase = (chunk - rs * 6) << 5;
        int r     = rs / 3;
        int s     = rs - r * 3;
        return combP + (b * CCOMB + cBase) * CHST + (y + r) * WP + s + tOff;
    };

    v8f acc00 = {}, acc01 = {}, acc10 = {}, acc11 = {};
    union PK { __bf16 h[8]; uint4 q; };

    auto issueStage = [&](int kc, int buf, PK& pv) {
        async_lds_b128(wl0[buf], wg0 + kc);
        async_lds_b128(wl1[buf], wg1 + kc);
        const __bf16* pb = patchBase(kc);
        #pragma unroll
        for (int i = 0; i < 8; ++i) pv.h[i] = pb[i * CHST];
        __builtin_prefetch(pb + 32 * CHST, 0, 3);
    };
    auto commitStage = [&](int buf, PK& pv) {
        *(uint4*)(&smem.st.P[buf][pl][kkb]) = pv.q;
    };
    auto compute = [&](int buf) {
        union Frag { v16bf v; unsigned u[8]; };
        Frag a0, a1, b0, b1;
        #pragma unroll
        for (int j = 0; j < 8; ++j) {
            int j0 = 2 * j;
            int kA = (j0 < 8) ? (j0 + (hi ? 8 : 0)) : (j0 + (hi ? 16 : 8));
            int kB = kA;
            a0.u[j] = *(const unsigned*)(&smem.st.W[buf][g * 32 + m][kA]);
            a1.u[j] = *(const unsigned*)(&smem.st.W[buf][g * 32 + 16 + m][kA]);
            b0.u[j] = *(const unsigned*)(&smem.st.P[buf][t * 32 + m][kB]);
            b1.u[j] = *(const unsigned*)(&smem.st.P[buf][t * 32 + 16 + m][kB]);
        }
        acc00 = __builtin_amdgcn_wmma_f32_16x16x32_bf16(false, a0.v, false, b0.v,
                                                        (short)0, acc00, false, false);
        acc01 = __builtin_amdgcn_wmma_f32_16x16x32_bf16(false, a0.v, false, b1.v,
                                                        (short)0, acc01, false, false);
        acc10 = __builtin_amdgcn_wmma_f32_16x16x32_bf16(false, a1.v, false, b0.v,
                                                        (short)0, acc10, false, false);
        acc11 = __builtin_amdgcn_wmma_f32_16x16x32_bf16(false, a1.v, false, b1.v,
                                                        (short)0, acc11, false, false);
    };

    // ---- prologue: stage chunk 0 -> buf0 ----
    {
        PK pv;
        issueStage(0, 0, pv);
        commitStage(0, pv);
        wait_asynccnt0();
        __syncthreads();
    }

    // ---- steady state: 26 iterations, 2 chunks each, unconditional ----
    for (int kc = 0; kc < KTOT - 64; kc += 64) {
        PK pv;
        issueStage(kc + 32, 1, pv);      // stage n+1 -> buf1
        compute(0);                      // compute n  from buf0
        commitStage(1, pv);
        wait_asynccnt0();
        __syncthreads();

        issueStage(kc + 64, 0, pv);      // stage n+2 -> buf0
        compute(1);                      // compute n+1 from buf1
        commitStage(0, pv);
        wait_asynccnt0();
        __syncthreads();
    }
    // ---- tail: chunks KTOT-64 (buf0) and KTOT-32 (buf1) ----
    {
        PK pv;
        issueStage(KTOT - 32, 1, pv);
        compute(0);
        commitStage(1, pv);
        wait_asynccnt0();
        __syncthreads();
        compute(1);
    }

    // ---- dump accumulators into overlaid Sout ----
    __syncthreads();
    #pragma unroll
    for (int j = 0; j < 8; ++j) {
        int row0 = g * 32 + j + hi * 8;
        int col0 = t * 32 + m;
        smem.Sout[row0     ][col0     ] = acc00[j];
        smem.Sout[row0     ][col0 + 16] = acc01[j];
        smem.Sout[row0 + 16][col0     ] = acc10[j];
        smem.Sout[row0 + 16][col0 + 16] = acc11[j];
    }
    __syncthreads();

    // ---- fused LSTM gating (native v_tanh_f32 path) ----
    #pragma unroll
    for (int i = 0; i < 8; ++i) {
        int e   = tid + i * 256;
        int ppl = e & 63;
        int cl  = e >> 6;
        int ch  = chb + cl;
        float xi = smem.Sout[cl      ][ppl] + bias[ch];
        float xf = smem.Sout[32 + cl ][ppl] + bias[CHID + ch];
        float xo = smem.Sout[64 + cl ][ppl] + bias[2 * CHID + ch];
        float xg = smem.Sout[96 + cl ][ppl] + bias[3 * CHID + ch];
        int idx  = ((b * CHID + ch) << 12) + (pixBase & 4095) + ppl;
        float c0 = c_cur[idx];
        float ig = sigmoid_hw(xi);
        float fg = sigmoid_hw(xf);
        float og = sigmoid_hw(xo);
        float gg = tanh_hw(xg);
        float cn = fg * c0 + ig * gg;
        float hn = og * tanh_hw(cn);
        c_next[idx] = cn;
        h_next[idx] = hn;
    }
}

// ---------------------------------------------------------------------------
extern "C" void kernel_launch(void* const* d_in, const int* in_sizes, int n_in,
                              void* d_out, int out_size, void* d_ws, size_t ws_size,
                              hipStream_t stream) {
    const float* x    = (const float*)d_in[0];
    const float* h    = (const float*)d_in[1];
    const float* c    = (const float*)d_in[2];
    const float* w    = (const float*)d_in[3];
    const float* bias = (const float*)d_in[4];

    __bf16* wt    = (__bf16*)d_ws;                        // 1.77 MB
    __bf16* combP = (__bf16*)((char*)d_ws + (2u << 20));  // 29.2 MB padded

    float* h_next = (float*)d_out;
    float* c_next = h_next + (size_t)B_ * CHID * 4096;

    prep_weights<<<dim3(OC, 9), 192, 0, stream>>>(w, wt);
    prep_comb<<<dim3((CHST + 255) / 256, CCOMB, B_), 256, 0, stream>>>(x, h, combP);
    convlstm_wmma<<<dim3(NPIX / 64, CHID / 32), 256, 0, stream>>>(
        wt, combP, c, bias, h_next, c_next);
}
